// CMA_Block_81853486727997
// MI455X (gfx1250) — matrix operand order repaired and checked by reading, not compiled, not run on hardware
//
#include <hip/hip_runtime.h>

typedef __attribute__((ext_vector_type(16))) _Float16 v16h;
typedef __attribute__((ext_vector_type(8)))  _Float16 v8h;
typedef __attribute__((ext_vector_type(8)))  float    v8f;
typedef __attribute__((ext_vector_type(4)))  unsigned int v4u;
typedef __attribute__((ext_vector_type(8)))  int v8i;
typedef __attribute__((ext_vector_type(4)))  int v4i;

union V16H { v16h v; v8h h[2]; };

#define NBATCH 8
#define NTOK   4096   // 64x64 tokens after avgpool

// xor-butterfly within 16-lane half via single ds_swizzle_b32 (imm pattern)
#define SWIZ_XOR_F(x, mask) \
  __int_as_float(__builtin_amdgcn_ds_swizzle(__float_as_int(x), (((mask) << 10) | 0x1F)))

// ---------------------------------------------------------------------------
// TDM: 2-D tile load (element size 2B) global -> LDS, per CDNA5 D# layout.
// 6-arg builtin: (g0 u32x4, g1 i32x8, g2 i32x4, g3 i32x4, i32x8, cpol)
// ---------------------------------------------------------------------------
__device__ __forceinline__ void tdm_load_2d(unsigned lds_off, const void* gaddr,
                                            unsigned td0, unsigned td1,
                                            unsigned tile0, unsigned tile1,
                                            unsigned long long stride0) {
  const unsigned long long ga = (unsigned long long)(uintptr_t)gaddr;
  v4u g0;
  g0[0] = 1u;                                   // count=1, user mode
  g0[1] = lds_off;                              // LDS byte address
  g0[2] = (unsigned)ga;                         // global_addr[31:0]
  g0[3] = (unsigned)((ga >> 32) & 0x1FFFFFFu)   // global_addr[56:32]
          | (2u << 30);                         // type = 2 (image)
  v8i g1;
  g1[0] = (int)(1u << 16);                      // data_size=1 -> 2 bytes
  g1[1] = (int)((td0 & 0xFFFFu) << 16);         // tensor_dim0[15:0] @ bit48
  g1[2] = (int)((td0 >> 16) | ((td1 & 0xFFFFu) << 16));
  g1[3] = (int)((td1 >> 16) | (tile0 << 16));   // tile_dim0 @ bit112
  g1[4] = (int)tile1;                           // tile_dim1; tile_dim2=0
  g1[5] = (int)(unsigned)(stride0 & 0xFFFFFFFFull);   // dim0_stride[31:0]
  g1[6] = (int)(unsigned)((stride0 >> 32) & 0xFFFFull);
  g1[7] = 0;                                    // dim1_stride unused (2-D)
  v4i z4 = {0, 0, 0, 0};
  v8i z8 = {0, 0, 0, 0, 0, 0, 0, 0};
  __builtin_amdgcn_tensor_load_to_lds(g0, g1, z4, z4, z8, 0);
}

// ---------------------------------------------------------------------------
// Kernel 1: avgpool(2x2) + 1x1-conv projections.
//   mode 0: rgb  -> Q [B][N][64] f16 (scaled by 1/8)
//   mode 1: freq -> K [B][N][64] f16 and Vt [B][64][N] f16
// ---------------------------------------------------------------------------
__global__ __launch_bounds__(128) void qkv_kernel(
    const float* __restrict__ rgb, const float* __restrict__ freq,
    const float* __restrict__ w_q, const float* __restrict__ b_q,
    const float* __restrict__ w_k, const float* __restrict__ b_k,
    const float* __restrict__ w_v, const float* __restrict__ b_v,
    _Float16* __restrict__ Q, _Float16* __restrict__ K,
    _Float16* __restrict__ Vt) {
  __shared__ float wA[4096];
  __shared__ float wB[4096];
  __shared__ float bA[64];
  __shared__ float bB[64];
  const int tid  = threadIdx.x;
  const int mode = blockIdx.z;
  const float* WA = (mode == 0) ? w_q : w_k;
  const float* BA = (mode == 0) ? b_q : b_k;
  for (int i = tid; i < 4096; i += 128) {
    wA[i] = WA[i];
    wB[i] = w_v[i];
  }
  if (tid < 64) { bA[tid] = BA[tid]; bB[tid] = b_v[tid]; }
  __syncthreads();

  const int b  = blockIdx.y;
  const int t  = blockIdx.x * 128 + tid;
  const int hd = t >> 6, wd = t & 63;
  const float* src = (mode == 0) ? rgb : freq;
  const float* sp  = src + (((size_t)b * 64) * 128 + 2 * hd) * 128 + 2 * wd;

  float x[64];
#pragma unroll
  for (int c = 0; c < 64; ++c) {
    const float* p = sp + (size_t)c * 16384;
    x[c] = 0.25f * (p[0] + p[1] + p[128] + p[129]);
  }

  __align__(16) _Float16 outr[64];
  const float qs = (mode == 0) ? 0.125f : 1.0f;  // fold softmax scale into Q
#pragma unroll 4
  for (int oc = 0; oc < 64; ++oc) {
    float a = bA[oc];
    const float* wr = &wA[oc * 64];
#pragma unroll
    for (int ic = 0; ic < 64; ++ic) a = fmaf(wr[ic], x[ic], a);
    outr[oc] = (_Float16)(a * qs);
  }
  _Float16* dst = ((mode == 0) ? Q : K) + ((size_t)b * NTOK + t) * 64;
#pragma unroll
  for (int i = 0; i < 8; ++i) *(v8h*)(dst + 8 * i) = *(const v8h*)(outr + 8 * i);

  if (mode == 1) {
#pragma unroll 4
    for (int oc = 0; oc < 64; ++oc) {
      float a = bB[oc];
      const float* wr = &wB[oc * 64];
#pragma unroll
      for (int ic = 0; ic < 64; ++ic) a = fmaf(wr[ic], x[ic], a);
      Vt[((size_t)b * 64 + oc) * NTOK + t] = (_Float16)a;
    }
  }
}

// ---------------------------------------------------------------------------
// Kernel 2: flash-attention. Block = 4 waves = 64 query rows of one batch.
// Wave 0 double-buffers 64-kv K/V tiles into LDS via TDM; all 4 waves share.
// 16 v_wmma per 64-kv iteration.  Writes z [B][N][64] f32.
// ---------------------------------------------------------------------------
__global__ __launch_bounds__(128) void attn_kernel(
    const _Float16* __restrict__ Q, const _Float16* __restrict__ K,
    const _Float16* __restrict__ Vt, float* __restrict__ Z) {
  __shared__ __align__(16) _Float16 kbuf[2][64][64];  // [buf][kv_local][d]
  __shared__ __align__(16) _Float16 vbuf[2][64][64];  // [buf][d][kv_local]
  __shared__ __align__(16) _Float16 pls[4][16][64];   // per-wave P staging

  const int tid  = threadIdx.x;
  const int wave = tid >> 5;
  const int lane = tid & 31;
  const int b       = blockIdx.x >> 6;                   // 64 blocks per batch
  const int rowbase = ((blockIdx.x & 63) << 6) + (wave << 4);
  const int m  = lane & 15;
  const int hi = lane >> 4;

  const _Float16* Qb = Q  + (size_t)b * NTOK * 64;
  const _Float16* Kb = K  + (size_t)b * NTOK * 64;
  const _Float16* Vb = Vt + (size_t)b * 64 * NTOK;
  float*          Zb = Z  + (size_t)b * NTOK * 64;

  // A-operand (16x32 f16): lane m holds k = 16*hi..16*hi+15
  const _Float16* qrow = Qb + (size_t)(rowbase + m) * 64;
  V16H a0, a1;
  a0.h[0] = *(const v8h*)(qrow + 16 * hi);
  a0.h[1] = *(const v8h*)(qrow + 8 + 16 * hi);
  a1.h[0] = *(const v8h*)(qrow + 32 + 16 * hi);
  a1.h[1] = *(const v8h*)(qrow + 40 + 16 * hi);

  v8f acc[4] = {};
  float mrow[8], lpart[8];
#pragma unroll
  for (int r = 0; r < 8; ++r) { mrow[r] = -1e30f; lpart[r] = 0.0f; }

  _Float16* pw = &pls[wave][0][0];
  const unsigned koff[2] = {(unsigned)(uintptr_t)&kbuf[0][0][0],
                            (unsigned)(uintptr_t)&kbuf[1][0][0]};
  const unsigned voff[2] = {(unsigned)(uintptr_t)&vbuf[0][0][0],
                            (unsigned)(uintptr_t)&vbuf[1][0][0]};

  if (wave == 0) {
    tdm_load_2d(koff[0], Kb, 64, NTOK, 64, 64, 64);        // K rows kv..kv+63
    tdm_load_2d(voff[0], Vb, NTOK, 64, 64, 64, NTOK);      // Vt cols kv..kv+63
  }

  int cur = 0;
  for (int kv = 0; kv < NTOK; kv += 64) {
    if (wave == 0) __builtin_amdgcn_s_wait_tensorcnt(0);
    __syncthreads();   // tile `cur` ready; everyone done with tile `cur^1`
    if (wave == 0 && kv + 64 < NTOK) {
      tdm_load_2d(koff[cur ^ 1], Kb + (size_t)(kv + 64) * 64, 64, NTOK, 64, 64, 64);
      tdm_load_2d(voff[cur ^ 1], Vb + (kv + 64), NTOK, 64, 64, 64, NTOK);
    }
    const _Float16* kc = &kbuf[cur][0][0];
    const _Float16* vc = &vbuf[cur][0][0];

    // ---- S = Q*K^T : four 16x16 tiles, contraction d=64 in 2 chunks ----
    v8f s[4];
    V16H bk;
#pragma unroll
    for (int tn = 0; tn < 4; ++tn) {
      const _Float16* kp = kc + (tn * 16 + m) * 64 + 16 * hi;
      v8f st = {};
      bk.v = *(const v16h*)(kp);
      st = __builtin_amdgcn_wmma_f32_16x16x32_f16(false, a0.v, false, bk.v, (short)0, st, false, false);
      bk.v = *(const v16h*)(kp + 32);
      st = __builtin_amdgcn_wmma_f32_16x16x32_f16(false, a1.v, false, bk.v, (short)0, st, false, false);
      s[tn] = st;
    }

    // ---- row max over 64 columns (4 local + 4 swizzle steps) ----
    float t0[8];
#pragma unroll
    for (int r = 0; r < 8; ++r)
      t0[r] = fmaxf(fmaxf(s[0][r], s[1][r]), fmaxf(s[2][r], s[3][r]));
#pragma unroll
    for (int r = 0; r < 8; ++r) {
      float t = t0[r];
      t = fmaxf(t, SWIZ_XOR_F(t, 1));
      t = fmaxf(t, SWIZ_XOR_F(t, 2));
      t = fmaxf(t, SWIZ_XOR_F(t, 4));
      t = fmaxf(t, SWIZ_XOR_F(t, 8));
      t0[r] = t;
    }

    // ---- online softmax; per-lane partial row-sum (no cross-lane here) ----
#pragma unroll
    for (int r = 0; r < 8; ++r) {
      const float nm = fmaxf(mrow[r], t0[r]);
      const float al = __expf(mrow[r] - nm);
      mrow[r] = nm;
      float psum = 0.0f;
#pragma unroll
      for (int tn = 0; tn < 4; ++tn) {
        const float p = __expf(s[tn][r] - nm);
        s[tn][r] = p;
        psum += p;
      }
      lpart[r] = lpart[r] * al + psum;
      acc[0][r] = acc[0][r] * al;
      acc[1][r] = acc[1][r] * al;
      acc[2][r] = acc[2][r] * al;
      acc[3][r] = acc[3][r] * al;
    }

    // ---- restage P 16x64 (C-layout -> A-layout) via per-wave LDS ----
#pragma unroll
    for (int tn = 0; tn < 4; ++tn) {
#pragma unroll
      for (int r = 0; r < 8; ++r)
        pw[(hi * 8 + r) * 64 + tn * 16 + m] = (_Float16)s[tn][r];
    }
    V16H pa0, pa1;
    pa0.h[0] = *(const v8h*)(pw + m * 64 + 16 * hi);
    pa0.h[1] = *(const v8h*)(pw + m * 64 + 8 + 16 * hi);
    pa1.h[0] = *(const v8h*)(pw + m * 64 + 32 + 16 * hi);
    pa1.h[1] = *(const v8h*)(pw + m * 64 + 40 + 16 * hi);

    // ---- O += P*V : contraction over 64 kv in 2 chunks, 4 d-tiles ----
#pragma unroll
    for (int nb = 0; nb < 4; ++nb) {
      const _Float16* vp = vc + (nb * 16 + m) * 64 + 16 * hi;
      bk.v = *(const v16h*)(vp);
      acc[nb] = __builtin_amdgcn_wmma_f32_16x16x32_f16(false, pa0.v, false, bk.v, (short)0, acc[nb], false, false);
      bk.v = *(const v16h*)(vp + 32);
      acc[nb] = __builtin_amdgcn_wmma_f32_16x16x32_f16(false, pa1.v, false, bk.v, (short)0, acc[nb], false, false);
    }
    cur ^= 1;
  }

  // ---- final cross-lane row-sum, normalize, write z[token][c] ----
#pragma unroll
  for (int r = 0; r < 8; ++r) {
    float t = lpart[r];
    t += SWIZ_XOR_F(t, 1);
    t += SWIZ_XOR_F(t, 2);
    t += SWIZ_XOR_F(t, 4);
    t += SWIZ_XOR_F(t, 8);
    const float inv = 1.0f / t;
    const int token = rowbase + hi * 8 + r;
    float* zr = Zb + (size_t)token * 64 + m;
    zr[0]  = acc[0][r] * inv;
    zr[16] = acc[1][r] * inv;
    zr[32] = acc[2][r] * inv;
    zr[48] = acc[3][r] * inv;
  }
}

// ---------------------------------------------------------------------------
// Kernel 3: bilinear 2x upsample + 1x1 conv + BN(affine) + LeakyReLU + residual
// ---------------------------------------------------------------------------
__global__ __launch_bounds__(256) void out_kernel(
    const float* __restrict__ rgb,
    const float* __restrict__ w_o, const float* __restrict__ b_o,
    const float* __restrict__ gamma, const float* __restrict__ beta,
    const float* __restrict__ mean, const float* __restrict__ var,
    const float* __restrict__ Z, float* __restrict__ out) {
  __shared__ float wos[4096];
  __shared__ float bos[64], invs[64], bets[64], mns[64];
  const int tid = threadIdx.x;
  for (int i = tid; i < 4096; i += 256) wos[i] = w_o[i];
  if (tid < 64) {
    bos[tid]  = b_o[tid];
    invs[tid] = gamma[tid] * rsqrtf(var[tid] + 1e-5f);
    bets[tid] = beta[tid];
    mns[tid]  = mean[tid];
  }
  __syncthreads();

  const int idx = blockIdx.x * 256 + tid;
  const int wo = idx & 127;
  const int ho = (idx >> 7) & 127;
  const int b  = idx >> 14;

  float sy = fmaxf(ho * 0.5f - 0.25f, 0.0f);
  int   y0 = (int)sy;
  float wy = sy - (float)y0;
  int   y1 = min(y0 + 1, 63);
  float sx = fmaxf(wo * 0.5f - 0.25f, 0.0f);
  int   x0 = (int)sx;
  float wx = sx - (float)x0;
  int   x1 = min(x0 + 1, 63);

  const float* Zb  = Z + (size_t)b * NTOK * 64;
  const float* z00 = Zb + (size_t)(y0 * 64 + x0) * 64;
  const float* z01 = Zb + (size_t)(y0 * 64 + x1) * 64;
  const float* z10 = Zb + (size_t)(y1 * 64 + x0) * 64;
  const float* z11 = Zb + (size_t)(y1 * 64 + x1) * 64;
  const float c00 = (1.0f - wy) * (1.0f - wx);
  const float c01 = (1.0f - wy) * wx;
  const float c10 = wy * (1.0f - wx);
  const float c11 = wy * wx;

  float zin[64];
#pragma unroll
  for (int c = 0; c < 64; ++c)
    zin[c] = c00 * z00[c] + c01 * z01[c] + c10 * z10[c] + c11 * z11[c];

#pragma unroll 4
  for (int oc = 0; oc < 64; ++oc) {
    float a = bos[oc];
    const float* wr = &wos[oc * 64];
#pragma unroll
    for (int ic = 0; ic < 64; ++ic) a = fmaf(wr[ic], zin[ic], a);
    float y = (a - mns[oc]) * invs[oc] + bets[oc];
    y = (y >= 0.0f) ? y : 0.2f * y;
    const size_t o = (((size_t)b * 64 + oc) * 128 + ho) * 128 + wo;
    out[o] = rgb[o] + y;
  }
}

// ---------------------------------------------------------------------------
extern "C" void kernel_launch(void* const* d_in, const int* in_sizes, int n_in,
                              void* d_out, int out_size, void* d_ws, size_t ws_size,
                              hipStream_t stream) {
  (void)in_sizes; (void)n_in; (void)out_size; (void)ws_size;
  const float* rgb  = (const float*)d_in[0];
  const float* freq = (const float*)d_in[1];
  const float* w_q  = (const float*)d_in[2];
  const float* b_q  = (const float*)d_in[3];
  const float* w_k  = (const float*)d_in[4];
  const float* b_k  = (const float*)d_in[5];
  const float* w_v  = (const float*)d_in[6];
  const float* b_v  = (const float*)d_in[7];
  const float* w_o  = (const float*)d_in[8];
  const float* b_o  = (const float*)d_in[9];
  const float* gmm  = (const float*)d_in[10];
  const float* bet  = (const float*)d_in[11];
  const float* mea  = (const float*)d_in[12];
  const float* var  = (const float*)d_in[13];
  float* out = (float*)d_out;

  const size_t elems = (size_t)NBATCH * NTOK * 64;
  _Float16* Q  = (_Float16*)d_ws;
  _Float16* K  = Q + elems;
  _Float16* Vt = K + elems;
  float*    Z  = (float*)(Vt + elems);   // total ws use: 20 MiB

  qkv_kernel<<<dim3(32, NBATCH, 2), 128, 0, stream>>>(
      rgb, freq, w_q, b_q, w_k, b_k, w_v, b_v, Q, K, Vt);

  // 512 blocks x 4 waves; each block = 64 query rows sharing TDM-staged K/V
  attn_kernel<<<512, 128, 0, stream>>>(Q, K, Vt, Z);

  out_kernel<<<512, 256, 0, stream>>>(
      rgb, w_o, b_o, gmm, bet, mea, var, Z, out);
}
